// MolGraphTransformer_3547642987146
// MI455X (gfx1250) — matrix operand order repaired and checked by reading, not compiled, not run on hardware
//
#include <hip/hip_runtime.h>
#include <hip/hip_bf16.h>

typedef __attribute__((ext_vector_type(16))) _Float16 v16h;
typedef __attribute__((ext_vector_type(8)))  float    v8f;
typedef __attribute__((ext_vector_type(4)))  float    v4f;
typedef __attribute__((ext_vector_type(4)))  int      v4i;

#define HID 128
#define NH  4
#define DH  32

#if __has_builtin(__builtin_amdgcn_global_load_async_to_lds_b128) && \
    __has_builtin(__builtin_amdgcn_s_wait_asynccnt)
#define HAVE_ASYNC_LDS 1
#else
#define HAVE_ASYNC_LDS 0
#endif

#if HAVE_ASYNC_LDS
typedef __attribute__((address_space(1))) v4i* as1_v4i_ptr;   // global
typedef __attribute__((address_space(3))) v4i* as3_v4i_ptr;   // LDS
__device__ __forceinline__ as1_v4i_ptr to_gv4i(const void* p) {
  return (as1_v4i_ptr)(uintptr_t)p;                            // same 64-bit value
}
__device__ __forceinline__ as3_v4i_ptr to_lv4i(const void* p) {
  // generic LDS pointer -> 32-bit LDS offset (low 32 bits of the address)
  return (as3_v4i_ptr)(uintptr_t)(unsigned)(uintptr_t)p;
}
#endif

// ---------------------------------------------------------------------------
// Workspace init: acc=0, ssum=0, smax=-inf
// ---------------------------------------------------------------------------
__global__ void init_ws_kernel(float* __restrict__ acc, float* __restrict__ ssum,
                               unsigned* __restrict__ smax, long accN, long sN) {
  long stride = (long)gridDim.x * blockDim.x;
  long base = (long)blockIdx.x * blockDim.x + threadIdx.x;
  for (long t = base; t < accN; t += stride) acc[t] = 0.0f;
  for (long t = base; t < sN; t += stride) { ssum[t] = 0.0f; smax[t] = 0xFF800000u; }
}

// ---------------------------------------------------------------------------
// Pack one 128x128 f32 weight matrix into f16 B-fragment order:
//   idx = ((ct*4 + kk)*32 + lane)*16 + i
//   k   = kk*32 + (lane>>4)*16 + i ,  c = ct*16 + (lane&15)
// so a wave's B fragment for (ct,kk) is one contiguous 32B chunk per lane.
// ---------------------------------------------------------------------------
__global__ __launch_bounds__(256) void pack_w_kernel(const float* __restrict__ W,
                                                     _Float16* __restrict__ Wpk) {
  int r = blockIdx.x * 256 + threadIdx.x;       // 0..16383  (grid = 64)
  int i    = r & 15;
  int lane = (r >> 4) & 31;
  int kk   = (r >> 9) & 3;
  int ct   = r >> 11;
  int k = kk * 32 + (lane >> 4) * 16 + i;
  int c = ct * 16 + (lane & 15);
  Wpk[r] = (_Float16)W[k * HID + c];
}

// ---------------------------------------------------------------------------
// Tiled GEMM: out[M,128] = act( scale(A)[M,128] @ W[128,128] + bias )
// f16 WMMA 16x16x32, f32 accumulate. Block = 8 waves x 16 rows = 128 rows.
// Each wave keeps its 4 A fragments in registers across all 8 column tiles
// (32 WMMAs / wave). Packed-f16 W staged into LDS once per block (async-to-LDS
// path on toolchains that expose the gfx1250 builtin).
// rowscale (optional): A[r,k] /= rowscale[r*4 + (k>>5)]  (softmax denominator)
// ---------------------------------------------------------------------------
__global__ __launch_bounds__(256) void gemm128_wmma(
    const float* __restrict__ A, const _Float16* __restrict__ Wpk,
    const float* __restrict__ bias, const float* __restrict__ rowscale,
    float* __restrict__ out, int M, int relu_flag)
{
  __shared__ __align__(64) char smem[32768];    // 128x128 f16, fragment order
  const int lane = threadIdx.x & 31;
  const int wave = threadIdx.x >> 5;            // 0..7 -> 16-row strip in block
  const int r16  = lane & 15;
  const int hi   = lane >> 4;

  // ---- stage packed W into LDS (32KB, 8 x b128 per thread) ----
  const char* wsrc = (const char*)Wpk;
#pragma unroll
  for (int it = 0; it < 8; ++it) {
    int off = (it * 256 + threadIdx.x) * 16;
#if HAVE_ASYNC_LDS
    __builtin_amdgcn_global_load_async_to_lds_b128(to_gv4i(wsrc + off),
                                                   to_lv4i(smem + off), 0, 0);
#else
    *(v4f*)(smem + off) = *(const v4f*)(wsrc + off);
#endif
  }
#if HAVE_ASYNC_LDS
  __builtin_amdgcn_s_wait_asynccnt(0);
#endif
  __syncthreads();

  // ---- A fragments: contiguous k = kk*32 + hi*16 + i (same order as packed W) ----
  const int rowbase = blockIdx.x * 128 + wave * 16;
  const int row  = rowbase + r16;
  const int rowc = row < M ? row : (M - 1);     // clamp -> EXEC all-ones for WMMA

  v16h af[4];
#pragma unroll
  for (int kk = 0; kk < 4; ++kk) {              // K = 4 x 32; each step = one head
    float s = 1.0f;
    if (rowscale) {
      float dnm = rowscale[rowc * NH + kk];
      s = (dnm > 0.0f) ? __builtin_amdgcn_rcpf(dnm) : 0.0f;
    }
    const float* ap = A + (long)rowc * HID + kk * 32 + hi * 16;
    v4f a0 = *(const v4f*)(ap);
    v4f a1 = *(const v4f*)(ap + 4);
    v4f a2 = *(const v4f*)(ap + 8);
    v4f a3 = *(const v4f*)(ap + 12);
#pragma unroll
    for (int i = 0; i < 4; ++i) {
      af[kk][i]      = (_Float16)(a0[i] * s);
      af[kk][4 + i]  = (_Float16)(a1[i] * s);
      af[kk][8 + i]  = (_Float16)(a2[i] * s);
      af[kk][12 + i] = (_Float16)(a3[i] * s);
    }
  }

  // ---- 8 column tiles x 4 K-steps of WMMA, B fragments from LDS ----
#pragma unroll
  for (int ct = 0; ct < 8; ++ct) {
    v8f c = {0.f, 0.f, 0.f, 0.f, 0.f, 0.f, 0.f, 0.f};
#pragma unroll
    for (int kk = 0; kk < 4; ++kk) {
      const v16h bf = *(const v16h*)(smem + ct * 4096 + kk * 1024 + lane * 32);
      c = __builtin_amdgcn_wmma_f32_16x16x32_f16(
          /*neg_a=*/false, af[kk], /*neg_b=*/false, bf,
          /*c_mod=*/(short)0, c, /*reuse_a=*/false, /*reuse_b=*/false);
    }
    // C tile: VGPR v holds row v + hi*8, col = lane&15 (ISA 7.12.2)
    const int col = ct * 16 + r16;
    const float bv = bias[col];
#pragma unroll
    for (int v = 0; v < 8; ++v) {
      int rr = rowbase + v + hi * 8;
      if (rr < M) {
        float o = c[v] + bv;
        if (relu_flag) o = fmaxf(o, 0.0f);
        out[(long)rr * HID + col] = o;
      }
    }
  }
}

// ---------------------------------------------------------------------------
// Float atomic max via int/uint punning (monotone-bits trick)
// ---------------------------------------------------------------------------
__device__ __forceinline__ void atomicMaxFloat(float* addr, float val) {
  if (!(val < 0.0f)) atomicMax((int*)addr, __float_as_int(val));
  else               atomicMin((unsigned int*)addr, __float_as_uint(val));
}

// ---------------------------------------------------------------------------
// Edge phase 1: score[e,h] = (K[src] . Q[dst])_h / sqrt(32); smax = segment max
// One wave32 per edge; coalesced 128B row gathers; butterfly shfl reduce.
// ---------------------------------------------------------------------------
__global__ __launch_bounds__(256) void edge_score_kernel(
    const float* __restrict__ Q, const float* __restrict__ K,
    const int* __restrict__ src, const int* __restrict__ dst,
    float* __restrict__ score, float* __restrict__ smax, int E)
{
  int e = (int)((blockIdx.x * 256u + threadIdx.x) >> 5);
  int lane = threadIdx.x & 31;
  if (e >= E) return;
  int s = src[e], d = dst[e];
  const float* kp = K + (long)s * HID;
  const float* qp = Q + (long)d * HID;
#pragma unroll
  for (int h = 0; h < NH; ++h) {
    float p = kp[h * DH + lane] * qp[h * DH + lane];
#pragma unroll
    for (int off = 16; off > 0; off >>= 1) p += __shfl_xor(p, off, 32);
    if (lane == 0) {
      float sc = p * 0.17677669529663689f;  // 1/sqrt(32)
      score[(long)e * NH + h] = sc;
      atomicMaxFloat(&smax[(long)d * NH + h], sc);
    }
  }
}

// ---------------------------------------------------------------------------
// Edge phase 2: w = exp(score - smax[dst]); ssum[dst] += w;
//               acc[dst] += w * V[src]   (normalization deferred to MLP GEMM)
// ---------------------------------------------------------------------------
__global__ __launch_bounds__(256) void edge_accum_kernel(
    const float* __restrict__ V, const int* __restrict__ src,
    const int* __restrict__ dst, const float* __restrict__ score,
    const float* __restrict__ smax, float* __restrict__ ssum,
    float* __restrict__ acc, int E)
{
  int e = (int)((blockIdx.x * 256u + threadIdx.x) >> 5);
  int lane = threadIdx.x & 31;
  if (e >= E) return;
  int s = src[e], d = dst[e];
#pragma unroll
  for (int h = 0; h < NH; ++h) {
    float w = __expf(score[(long)e * NH + h] - smax[(long)d * NH + h]);
    if (lane == 0) atomicAdd(&ssum[(long)d * NH + h], w);
    float val = w * V[(long)s * HID + h * DH + lane];
    atomicAdd(&acc[(long)d * HID + h * DH + lane], val);
  }
}

// ---------------------------------------------------------------------------
// Residual + LayerNorm: out = LN(h + t2) * gamma + beta. One wave per node.
// ---------------------------------------------------------------------------
__global__ __launch_bounds__(256) void resid_ln_kernel(
    const float* __restrict__ h, const float* __restrict__ t2,
    const float* __restrict__ gamma, const float* __restrict__ beta,
    float* __restrict__ out, int N)
{
  int n = (int)((blockIdx.x * 256u + threadIdx.x) >> 5);
  int lane = threadIdx.x & 31;
  if (n >= N) return;
  float x[4];
  float sum = 0.0f;
#pragma unroll
  for (int j = 0; j < 4; ++j) {
    long idx = (long)n * HID + j * 32 + lane;
    x[j] = h[idx] + t2[idx];
    sum += x[j];
  }
#pragma unroll
  for (int off = 16; off > 0; off >>= 1) sum += __shfl_xor(sum, off, 32);
  float mu = sum * (1.0f / 128.0f);
  float vs = 0.0f;
#pragma unroll
  for (int j = 0; j < 4; ++j) { float dv = x[j] - mu; vs += dv * dv; }
#pragma unroll
  for (int off = 16; off > 0; off >>= 1) vs += __shfl_xor(vs, off, 32);
  float rstd = rsqrtf(vs * (1.0f / 128.0f) + 1e-5f);
#pragma unroll
  for (int j = 0; j < 4; ++j) {
    int cidx = j * 32 + lane;
    out[(long)n * HID + cidx] = (x[j] - mu) * rstd * gamma[cidx] + beta[cidx];
  }
}

// ---------------------------------------------------------------------------
extern "C" void kernel_launch(void* const* d_in, const int* in_sizes, int n_in,
                              void* d_out, int out_size, void* d_ws, size_t ws_size,
                              hipStream_t stream) {
  const float* h   = (const float*)d_in[0];
  const int*   src = (const int*)d_in[1];
  const int*   dst = (const int*)d_in[2];
  const float* Wq = (const float*)d_in[3],  *bq = (const float*)d_in[4];
  const float* Wk = (const float*)d_in[5],  *bk = (const float*)d_in[6];
  const float* Wv = (const float*)d_in[7],  *bv = (const float*)d_in[8];
  const float* W1 = (const float*)d_in[9],  *b1 = (const float*)d_in[10];
  const float* W2 = (const float*)d_in[11], *b2 = (const float*)d_in[12];
  const float* gamma = (const float*)d_in[13], *beta = (const float*)d_in[14];
  float* out = (float*)d_out;

  const int N = in_sizes[0] / HID;   // 50000
  const int E = in_sizes[1];         // 800000
  const long nh = (long)N * HID;

  char* ws = (char*)d_ws;
  float* Qb    = (float*)ws; ws += nh * sizeof(float);
  float* Kb    = (float*)ws; ws += nh * sizeof(float);
  float* Vb    = (float*)ws; ws += nh * sizeof(float);
  float* acc   = (float*)ws; ws += nh * sizeof(float);
  float* score = (float*)ws; ws += (long)E * NH * sizeof(float);
  float* smax  = (float*)ws; ws += (long)N * NH * sizeof(float);
  float* ssum  = (float*)ws; ws += (long)N * NH * sizeof(float);
  _Float16* Wpk = (_Float16*)ws; ws += 5L * HID * HID * sizeof(_Float16);

  init_ws_kernel<<<1024, 256, 0, stream>>>(acc, ssum, (unsigned*)smax, nh, (long)N * NH);

  // pack the five 128x128 weight matrices into f16 fragment order (once/launch)
  pack_w_kernel<<<64, 256, 0, stream>>>(Wq, Wpk + 0 * 16384);
  pack_w_kernel<<<64, 256, 0, stream>>>(Wk, Wpk + 1 * 16384);
  pack_w_kernel<<<64, 256, 0, stream>>>(Wv, Wpk + 2 * 16384);
  pack_w_kernel<<<64, 256, 0, stream>>>(W1, Wpk + 3 * 16384);
  pack_w_kernel<<<64, 256, 0, stream>>>(W2, Wpk + 4 * 16384);

  const int gb = (N + 127) / 128;    // 128-row blocks
  gemm128_wmma<<<gb, 256, 0, stream>>>(h, Wpk + 0 * 16384, bq, nullptr, Qb, N, 0);
  gemm128_wmma<<<gb, 256, 0, stream>>>(h, Wpk + 1 * 16384, bk, nullptr, Kb, N, 0);
  gemm128_wmma<<<gb, 256, 0, stream>>>(h, Wpk + 2 * 16384, bv, nullptr, Vb, N, 0);

  const int eb = (E + 7) / 8;        // 8 waves (edges) per 256-thread block
  edge_score_kernel<<<eb, 256, 0, stream>>>(Qb, Kb, src, dst, score, smax, E);
  edge_accum_kernel<<<eb, 256, 0, stream>>>(Vb, src, dst, score, smax, ssum, acc, E);

  // MLP: t1 = relu((acc/ssum) @ W1 + b1)  (normalization fused into A load)
  gemm128_wmma<<<gb, 256, 0, stream>>>(acc, Wpk + 3 * 16384, b1, ssum, Qb, N, 1);
  gemm128_wmma<<<gb, 256, 0, stream>>>(Qb, Wpk + 4 * 16384, b2, nullptr, Kb, N, 1);

  const int nb = (N + 7) / 8;
  resid_ln_kernel<<<nb, 256, 0, stream>>>(h, Kb, gamma, beta, out, N);
}
